// FlashRWAttention_82042465288725
// MI455X (gfx1250) — matrix operand order repaired and checked by reading, not compiled, not run on hardware
//
#include <hip/hip_runtime.h>

// ---------------------------------------------------------------------------
// CDNA5 / gfx1250 fused RoPE + causal MQA attention.
// bf16 WMMA for all matmuls, async global->LDS double-buffering in attention.
// ---------------------------------------------------------------------------

typedef __bf16 bf16_t;
typedef __bf16 v16bf __attribute__((ext_vector_type(16)));
typedef __bf16 v8bf  __attribute__((ext_vector_type(8)));
typedef float  v8f   __attribute__((ext_vector_type(8)));
typedef float  v16f  __attribute__((ext_vector_type(16)));

#define WMMA_BF16(a, b, c) \
  __builtin_amdgcn_wmma_f32_16x16x32_bf16(false, (a), false, (b), (short)0, (c), false, false)

constexpr int BB   = 2;
constexpr int S    = 2048;
constexpr int H    = 32;
constexpr int D    = 64;
constexpr int HID  = 2048;          // H * D
constexpr int T    = BB * S;        // 4096
constexpr int NQKV = (H + 2) * D;   // 2176
constexpr float SCALE = 0.125f;     // D^-0.5

// --- gfx1250 async global->LDS copy (16B per lane), ASYNCcnt-tracked --------
__device__ __forceinline__ void async_cp16(const void* g, const void* lds_generic) {
  // Generic shared-pointer low 32 bits == logical LDS byte offset (ISA 10.2).
  unsigned           loff = (unsigned)(size_t)lds_generic;
  unsigned long long ga   = (unsigned long long)(size_t)g;
  asm volatile("global_load_async_to_lds_b128 %0, %1, off"
               :: "v"(loff), "v"(ga) : "memory");
}
__device__ __forceinline__ void wait_async_le16() {
  asm volatile("s_wait_asynccnt 0x10" ::: "memory");
}
__device__ __forceinline__ void wait_async_0() {
  asm volatile("s_wait_asynccnt 0x0" ::: "memory");
}

// --- fragment loaders (contiguous k order: element s <-> k = 16*(lane/16) + s), lane-adjusted pointers ---
__device__ __forceinline__ v16bf loadA_f32(const float* p) {
  v8f lo = *(const v8f*)p;
  v8f hi = *(const v8f*)(p + 8);
  v16bf a;
#pragma unroll
  for (int i = 0; i < 8; ++i) { a[i] = (bf16_t)lo[i]; a[i + 8] = (bf16_t)hi[i]; }
  return a;
}
__device__ __forceinline__ v16bf loadB_f32(const float* p) {
  v16f w = *(const v16f*)p;
  v16bf b;
#pragma unroll
  for (int i = 0; i < 16; ++i) b[i] = (bf16_t)w[i];
  return b;
}
__device__ __forceinline__ v16bf loadA_bf16(const bf16_t* p) {
  v8bf lo = *(const v8bf*)p;
  v8bf hi = *(const v8bf*)(p + 8);
  v16bf a;
#pragma unroll
  for (int i = 0; i < 8; ++i) { a[i] = lo[i]; a[i + 8] = hi[i]; }
  return a;
}
__device__ __forceinline__ v16bf loadB_bf16_pair(const bf16_t* p) {  // 16 contiguous
  v8bf lo = *(const v8bf*)p;
  v8bf hi = *(const v8bf*)(p + 8);
  v16bf b;
#pragma unroll
  for (int i = 0; i < 8; ++i) { b[i] = lo[i]; b[i + 8] = hi[i]; }
  return b;
}

// ---------------------------------------------------------------------------
// Kernel 1: QKV projection.  qkv[T x 2176] (bf16) = hs @ w_qkv^T.
// One wave -> 32x64 output tile (8 wmma / K-step), K loop 2048/32.
// ---------------------------------------------------------------------------
__global__ __launch_bounds__(32)
void qkv_gemm(const float* __restrict__ hs, const float* __restrict__ wqkv,
              bf16_t* __restrict__ qkv) {
  const int lane = threadIdx.x;
  const int l15  = lane & 15;
  const int m0   = blockIdx.x * 32;
  const int n0   = blockIdx.y * 64;
  const int kbA  = (lane < 16) ? 0 : 16;
  const int kbB  = (lane < 16) ? 0 : 16;
  const float* pa0 = hs + (size_t)(m0 + l15) * HID + kbA;
  const float* pa1 = pa0 + (size_t)16 * HID;
  const float* pb0 = wqkv + (size_t)(n0 + l15) * HID + kbB;

  v8f acc[2][4] = {};
  for (int k0 = 0; k0 < HID; k0 += 32) {
    __builtin_prefetch(pa0 + k0 + 64, 0, 1);
    __builtin_prefetch(pb0 + k0 + 64, 0, 1);
    v16bf a0 = loadA_f32(pa0 + k0);
    v16bf a1 = loadA_f32(pa1 + k0);
#pragma unroll
    for (int nt = 0; nt < 4; ++nt) {
      v16bf b = loadB_f32(pb0 + (size_t)nt * 16 * HID + k0);
      acc[0][nt] = WMMA_BF16(a0, b, acc[0][nt]);
      acc[1][nt] = WMMA_BF16(a1, b, acc[1][nt]);
    }
  }

  const int rowb = (lane < 16) ? 0 : 8;
#pragma unroll
  for (int mt = 0; mt < 2; ++mt)
#pragma unroll
    for (int nt = 0; nt < 4; ++nt)
#pragma unroll
      for (int r = 0; r < 8; ++r)
        qkv[(size_t)(m0 + mt * 16 + rowb + r) * NQKV + n0 + nt * 16 + l15] =
            (bf16_t)acc[mt][nt][r];
}

// ---------------------------------------------------------------------------
// Kernel 2: RoPE on q/k, and V stored TRANSPOSED: vt[b, d, s] so the P.V
// B-fragments become contiguous loads.  units 0..31 = q heads, 32 = k, 33 = v.
// ---------------------------------------------------------------------------
__global__ void rope_split(const bf16_t* __restrict__ qkv,
                           const float* __restrict__ cosT,
                           const float* __restrict__ sinT,
                           bf16_t* __restrict__ q, bf16_t* __restrict__ k,
                           bf16_t* __restrict__ vt) {
  int idx = blockIdx.x * blockDim.x + threadIdx.x;
  if (idx >= T * 34 * 32) return;
  const int d = idx & 31;
  const int u = (idx >> 5) % 34;
  const int t = idx / (34 * 32);
  const float c = cosT[t * 32 + d];
  const float s = sinT[t * 32 + d];
  if (u < 32) {
    float x1 = (float)qkv[(size_t)t * NQKV + u * 64 + d];
    float x2 = (float)qkv[(size_t)t * NQKV + u * 64 + d + 32];
    q[(size_t)t * HID + u * 64 + d]      = (bf16_t)(x1 * c - x2 * s);
    q[(size_t)t * HID + u * 64 + d + 32] = (bf16_t)(x2 * c + x1 * s);
  } else if (u == 32) {
    float x1 = (float)qkv[(size_t)t * NQKV + 2048 + d];
    float x2 = (float)qkv[(size_t)t * NQKV + 2048 + d + 32];
    k[(size_t)t * D + d]      = (bf16_t)(x1 * c - x2 * s);
    k[(size_t)t * D + d + 32] = (bf16_t)(x2 * c + x1 * s);
  } else {
    const int bb = t >> 11, ss = t & (S - 1);
    vt[((size_t)bb * D + d) * S + ss]      = qkv[(size_t)t * NQKV + 2112 + d];
    vt[((size_t)bb * D + d + 32) * S + ss] = qkv[(size_t)t * NQKV + 2112 + d + 32];
  }
}

// ---------------------------------------------------------------------------
// Kernel 3: causal MQA flash attention, one wave per (b, h, 16-row tile).
// K tile (32x64) and Vt tile (64x32) are double-buffered in LDS and filled
// with global_load_async_to_lds_b128; compute overlaps the next tile's copy.
// ---------------------------------------------------------------------------
__global__ __launch_bounds__(32)
void flash_attn(const bf16_t* __restrict__ q, const bf16_t* __restrict__ kc,
                const bf16_t* __restrict__ vt, bf16_t* __restrict__ attn) {
  __shared__ bf16_t Kl[2][32 * 72];   // rows = kv, cols = feature, ld 72 (pad)
  __shared__ bf16_t Vl[2][64 * 40];   // rows = d,  cols = kv,      ld 40 (pad)
  __shared__ bf16_t Pl[16 * 32];      // P tile, row-major, ld 32

  const int lane = threadIdx.x;
  const int l15  = lane & 15;
  const int half = (lane < 16) ? 0 : 1;
  const int m0   = blockIdx.x * 16;
  const int h    = blockIdx.y;
  const int b    = blockIdx.z;
  const int rowb = half * 8;

  // Q fragments for feature chunks 0..31 and 32..63.
  v16bf qa[2];
  {
    const bf16_t* qp = q + (size_t)(b * S + m0 + l15) * HID + h * D + half * 16;
#pragma unroll
    for (int c = 0; c < 2; ++c) {
      v8bf lo = *(const v8bf*)(qp + c * 32);
      v8bf hi = *(const v8bf*)(qp + c * 32 + 8);
#pragma unroll
      for (int i = 0; i < 8; ++i) { qa[c][i] = lo[i]; qa[c][i + 8] = hi[i]; }
    }
  }

  // Issue async fill of one KV tile (16 x 16B per lane = K 4KB + Vt 4KB).
  auto stage = [&](int kv0, int bufi) {
    const bf16_t* kg = kc + (size_t)(b * S + kv0 + lane) * D;
    bf16_t* kl = &Kl[bufi][lane * 72];
#pragma unroll
    for (int c = 0; c < 8; ++c) async_cp16(kg + c * 8, kl + c * 8);
#pragma unroll
    for (int j = 0; j < 2; ++j) {
      const bf16_t* vg = vt + (size_t)(b * D + j * 32 + lane) * S + kv0;
      bf16_t* vl = &Vl[bufi][(j * 32 + lane) * 40];
#pragma unroll
      for (int c = 0; c < 4; ++c) async_cp16(vg + c * 8, vl + c * 8);
    }
  };

  float runm[8], runl[8];
#pragma unroll
  for (int r = 0; r < 8; ++r) { runm[r] = -3.0e38f; runl[r] = 0.0f; }
  v8f acc[4] = {};

  const int kvend = m0 + 16;            // exclusive causal bound for this tile
  stage(0, 0);

  for (int kv0 = 0; kv0 < kvend; kv0 += 32) {
    const int cur = (kv0 >> 5) & 1;
    if (kv0 + 32 < kvend) {             // prefetch next tile, drain current
      stage(kv0 + 32, cur ^ 1);
      wait_async_le16();
    } else {
      wait_async_0();
    }

    // ---- scores: two 16x16 C tiles over kv columns [kv0, kv0+32) ----
    v8f sc[2] = {};
#pragma unroll
    for (int ct = 0; ct < 2; ++ct) {
      const bf16_t* kp = &Kl[cur][(ct * 16 + l15) * 72 + half * 16];
#pragma unroll
      for (int c = 0; c < 2; ++c) {
        v16bf bb = loadB_bf16_pair(kp + c * 32);
        sc[ct] = WMMA_BF16(qa[c], bb, sc[ct]);
      }
    }

    // ---- online softmax (per-row; rows live in 16-lane halves) ----
    float p0[8], p1[8], alpha[8];
#pragma unroll
    for (int r = 0; r < 8; ++r) {
      const int m_abs = m0 + rowb + r;
      float s0 = sc[0][r] * SCALE;
      float s1 = sc[1][r] * SCALE;
      if (kv0 + l15 > m_abs)      s0 = -1e30f;
      if (kv0 + 16 + l15 > m_abs) s1 = -1e30f;
      float tmax = fmaxf(s0, s1);
#pragma unroll
      for (int w = 8; w >= 1; w >>= 1) tmax = fmaxf(tmax, __shfl_xor(tmax, w, 16));
      const float newm = fmaxf(runm[r], tmax);
      const float a    = __expf(runm[r] - newm);
      runm[r] = newm;
      float e0 = __expf(s0 - newm);
      float e1 = __expf(s1 - newm);
      float tsum = e0 + e1;
#pragma unroll
      for (int w = 8; w >= 1; w >>= 1) tsum += __shfl_xor(tsum, w, 16);
      runl[r]  = runl[r] * a + tsum;
      alpha[r] = a;
      p0[r] = e0; p1[r] = e1;
    }
#pragma unroll
    for (int dt = 0; dt < 4; ++dt)
#pragma unroll
      for (int r = 0; r < 8; ++r) acc[dt][r] *= alpha[r];

    // ---- relayout P (C layout -> A layout) through 1KB LDS tile ----
#pragma unroll
    for (int r = 0; r < 8; ++r) {
      Pl[(rowb + r) * 32 + l15]      = (bf16_t)p0[r];
      Pl[(rowb + r) * 32 + 16 + l15] = (bf16_t)p1[r];
    }
    __asm__ volatile("s_wait_dscnt 0" ::: "memory");
    v16bf pa = loadA_bf16(&Pl[l15 * 32 + half * 16]);

    // ---- P.V: contiguous Vt fragments from LDS ----
#pragma unroll
    for (int dt = 0; dt < 4; ++dt) {
      v16bf vb = loadB_bf16_pair(&Vl[cur][(dt * 16 + l15) * 40 + half * 16]);
      acc[dt] = WMMA_BF16(pa, vb, acc[dt]);
    }
  }

  // ---- normalize and write attn (T x HID, bf16) ----
#pragma unroll
  for (int dt = 0; dt < 4; ++dt)
#pragma unroll
    for (int r = 0; r < 8; ++r) {
      float o = acc[dt][r] / runl[r];
      attn[(size_t)(b * S + m0 + rowb + r) * HID + h * D + dt * 16 + l15] = (bf16_t)o;
    }
}

// ---------------------------------------------------------------------------
// Kernel 4: output projection.  out[T x 2048] (f32) = attn(bf16) @ w_dense^T.
// One wave -> 32x64 tile, 8 wmma per K-step.
// ---------------------------------------------------------------------------
__global__ __launch_bounds__(32)
void dense_gemm(const bf16_t* __restrict__ attn, const float* __restrict__ wd,
                float* __restrict__ out) {
  const int lane = threadIdx.x;
  const int l15  = lane & 15;
  const int m0   = blockIdx.x * 32;
  const int n0   = blockIdx.y * 64;
  const int kbA  = (lane < 16) ? 0 : 16;
  const int kbB  = (lane < 16) ? 0 : 16;
  const bf16_t* pa0 = attn + (size_t)(m0 + l15) * HID + kbA;
  const bf16_t* pa1 = pa0 + (size_t)16 * HID;
  const float*  pb0 = wd + (size_t)(n0 + l15) * HID + kbB;

  v8f acc[2][4] = {};
  for (int k0 = 0; k0 < HID; k0 += 32) {
    __builtin_prefetch(pb0 + k0 + 64, 0, 1);
    v16bf a0 = loadA_bf16(pa0 + k0);
    v16bf a1 = loadA_bf16(pa1 + k0);
#pragma unroll
    for (int nt = 0; nt < 4; ++nt) {
      v16bf b = loadB_f32(pb0 + (size_t)nt * 16 * HID + k0);
      acc[0][nt] = WMMA_BF16(a0, b, acc[0][nt]);
      acc[1][nt] = WMMA_BF16(a1, b, acc[1][nt]);
    }
  }

  const int rowb = (lane < 16) ? 0 : 8;
#pragma unroll
  for (int mt = 0; mt < 2; ++mt)
#pragma unroll
    for (int nt = 0; nt < 4; ++nt)
#pragma unroll
      for (int r = 0; r < 8; ++r)
        out[(size_t)(m0 + mt * 16 + rowb + r) * HID + n0 + nt * 16 + l15] =
            acc[mt][nt][r];
}

// ---------------------------------------------------------------------------
extern "C" void kernel_launch(void* const* d_in, const int* in_sizes, int n_in,
                              void* d_out, int out_size, void* d_ws, size_t ws_size,
                              hipStream_t stream) {
  const float* hs   = (const float*)d_in[0];
  const float* cosT = (const float*)d_in[1];
  const float* sinT = (const float*)d_in[2];
  const float* wqkv = (const float*)d_in[3];
  const float* wd   = (const float*)d_in[4];
  float* out = (float*)d_out;

  char* ws = (char*)d_ws;
  bf16_t* qkv = (bf16_t*)ws; ws += (size_t)T * NQKV * sizeof(bf16_t);  // 17.8 MB
  bf16_t* qb  = (bf16_t*)ws; ws += (size_t)T * HID  * sizeof(bf16_t);  // 16.8 MB
  bf16_t* kb  = (bf16_t*)ws; ws += (size_t)T * D    * sizeof(bf16_t);  //  0.5 MB
  bf16_t* vtb = (bf16_t*)ws; ws += (size_t)BB * D * S * sizeof(bf16_t);//  0.5 MB
  bf16_t* ab  = (bf16_t*)ws;                                           // 16.8 MB

  qkv_gemm<<<dim3(T / 32, NQKV / 64), 32, 0, stream>>>(hs, wqkv, qkv);

  const int nrope = T * 34 * 32;
  rope_split<<<(nrope + 255) / 256, 256, 0, stream>>>(qkv, cosT, sinT, qb, kb, vtb);

  flash_attn<<<dim3(S / 16, H, BB), 32, 0, stream>>>(qb, kb, vtb, ab);

  dense_gemm<<<dim3(T / 32, HID / 64), 32, 0, stream>>>(ab, wd, out);
}
